// HS_MSA_35579509080462
// MI455X (gfx1250) — matrix operand order repaired and checked by reading, not compiled, not run on hardware
//
#include <hip/hip_runtime.h>
#include <hip/hip_bf16.h>

typedef __attribute__((ext_vector_type(16))) _Float16 v16h;
typedef __attribute__((ext_vector_type(8)))  _Float16 v8h;
typedef __attribute__((ext_vector_type(8)))  float    v8f;
typedef int v4i_ __attribute__((vector_size(16)));

#define AS1 __attribute__((address_space(1)))
#define AS3 __attribute__((address_space(3)))

#define B_    32
#define H_    32
#define W_    40
#define DIM_  224
#define N_    1280          // H_*W_
#define HEADS_ 8
#define DH_   28
#define INNER_ 224
#define L_    8960          // W_*DIM_
#define DM_   32            // D_MODEL
#define DSTATE_ 16
#define DI_   64            // D_INNER
#define RS_   0.70710678118654752440f
#define MROWS_ (B_*L_)      // 286720 mamba token rows
#define QROWS_ (B_*N_)      // 40960 attention token rows
#define WIN_TOK_ 64
#define NWIN_ (B_*6)        // 192 windows per subband
#define TROWS_ (NWIN_*WIN_TOK_)  // 12288

// ---- CDNA5 async global->LDS (ASYNCcnt) with safe fallback -------------------
__device__ __forceinline__ void async_copy16(const _Float16* g, _Float16* l) {
#if __has_builtin(__builtin_amdgcn_global_load_async_to_lds_b128)
  __builtin_amdgcn_global_load_async_to_lds_b128(
      (AS1 v4i_*)(void*)g, (AS3 v4i_*)(void*)l, 0, 0);
#else
  *(v8h*)l = *(const v8h*)g;
#endif
}
__device__ __forceinline__ void wait_async() {
#if __has_builtin(__builtin_amdgcn_s_wait_asynccnt)
  __builtin_amdgcn_s_wait_asynccnt(0);
#endif
}

// ---------------------------------------------------------------- cast f32->f16
__global__ __launch_bounds__(256) void k_cast_f16(const float* __restrict__ s,
                                                  _Float16* __restrict__ d, int n) {
  int i = blockIdx.x * 256 + threadIdx.x;
  if (i < n) d[i] = (_Float16)s[i];
}

// ---------------------------------------------------------------- WMMA GEMM
// C[M,N](f32) = A[M,K](f16,row-major,lda=K) * B[K,N](f16,row-major,ldb)
// M % 128 == 0, K % 32 == 0 required; N arbitrary (guarded).
// Block = 128 threads (4 waves). Block tile 128x64, wave tile 32x64, K-step 32.
__global__ __launch_bounds__(128) void k_gemm(const _Float16* __restrict__ A,
                                              const _Float16* __restrict__ B,
                                              float* __restrict__ C,
                                              int M, int N, int K, int ldb, int ldc) {
  __shared__ __align__(16) _Float16 As[128 * 40];  // As[r][k], stride 40
  __shared__ __align__(16) _Float16 Bs[64 * 40];   // Bs[n][k] (transposed), stride 40
  const int tid  = threadIdx.x;
  const int lane = tid & 31;
  const int wave = tid >> 5;
  const int m0 = blockIdx.y * 128;
  const int n0 = blockIdx.x * 64;

  v8f acc[2][4] = {};

  for (int k0 = 0; k0 < K; k0 += 32) {
    // stage A tile 128x32: thread t stages one full row (32 halves = 4x16B, async)
    {
      const int r = tid;
      const _Float16* gp = A + (size_t)(m0 + r) * K + k0;
      _Float16* lp = &As[r * 40];
      #pragma unroll
      for (int q = 0; q < 4; ++q) async_copy16(gp + q * 8, lp + q * 8);
      if (k0 + 32 < K)  // hint next K tile into cache (global_prefetch_b8)
        __builtin_prefetch(gp + 32, 0, 0);
    }
    // stage B tile 32x64 transposed: thread t -> col n=t>>1, k-half (t&1)*16
    {
      int n  = tid >> 1;
      int kh = (tid & 1) * 16;
      int gn = n0 + n;
      if (gn < N) {
        #pragma unroll
        for (int kk = 0; kk < 16; ++kk)
          Bs[n * 40 + kh + kk] = B[(size_t)(k0 + kh + kk) * ldb + gn];
      } else {
        #pragma unroll
        for (int kk = 0; kk < 16; ++kk) Bs[n * 40 + kh + kk] = (_Float16)0.f;
      }
    }
    wait_async();
    __syncthreads();

    // A fragments: lane holds row m=lane&15; K = kb+{0..7} and kb+16+{0..7}, kb=(lane>=16)*8
    const int kb  = (lane >> 4) * 8;
    const int kb2 = (lane >> 4) * 16;
    v16h afrag[2];
    #pragma unroll
    for (int mt = 0; mt < 2; ++mt) {
      int mrow = wave * 32 + mt * 16 + (lane & 15);
      v8h a0 = *(const v8h*)&As[mrow * 40 + kb];
      v8h a1 = *(const v8h*)&As[mrow * 40 + kb + 16];
      #pragma unroll
      for (int e = 0; e < 8; ++e) { afrag[mt][e] = a0[e]; afrag[mt][e + 8] = a1[e]; }
    }
    // B fragments: lane holds col n=lane&15; K = kb+{0..7} and kb+16+{0..7}
    #pragma unroll
    for (int nt = 0; nt < 4; ++nt) {
      int nloc = nt * 16 + (lane & 15);
      v16h bfrag;
      v8h b0 = *(const v8h*)&Bs[nloc * 40 + kb];
      v8h b1 = *(const v8h*)&Bs[nloc * 40 + kb + 16];
      #pragma unroll
      for (int e = 0; e < 8; ++e) { bfrag[e] = b0[e]; bfrag[e + 8] = b1[e]; }
      acc[0][nt] = __builtin_amdgcn_wmma_f32_16x16x32_f16(
          false, afrag[0], false, bfrag, (short)0, acc[0][nt], false, false);
      acc[1][nt] = __builtin_amdgcn_wmma_f32_16x16x32_f16(
          false, afrag[1], false, bfrag, (short)0, acc[1][nt], false, false);
    }
    __syncthreads();
  }

  // store: lane holds col=lane&15 within n-tile; rows r + (lane>=16)*8, r=0..7
  const int cbase = n0 + (lane & 15);
  #pragma unroll
  for (int mt = 0; mt < 2; ++mt) {
    const int rbase = m0 + wave * 32 + mt * 16 + (lane >> 4) * 8;
    #pragma unroll
    for (int nt = 0; nt < 4; ++nt) {
      int col = cbase + nt * 16;
      if (col < N) {
        #pragma unroll
        for (int r = 0; r < 8; ++r)
          C[(size_t)(rbase + r) * ldc + col] = acc[mt][nt][r];
      }
    }
  }
}

// ---------------------------------------------------------------- channel attention
// qb: (B,N,224)  kvb: (B,N,448)  xin: original x  xout: x + attn  (both (B,N,224))
__global__ __launch_bounds__(256) void k_chan_attn(const float* __restrict__ qb,
                                                   const float* __restrict__ kvb,
                                                   const float* __restrict__ xin,
                                                   float* __restrict__ xout) {
  const int bh = blockIdx.x;
  const int b = bh >> 3, h = bh & 7;
  const int tid = threadIdx.x;
  const float scale = 0.18898223650461363f;  // 28^-0.5
  __shared__ float kk[8][28], qq[8][28], vv[8][28];
  __shared__ float S[28][29];
  __shared__ float ksq[28], qsq[28];

  const int p[4] = { tid, tid + 256, tid + 512, tid + 768 };
  float g[4] = {0.f, 0.f, 0.f, 0.f};
  float sq = 0.f;

  for (int nb = 0; nb < N_; nb += 8) {
    for (int idx = tid; idx < 8 * 28; idx += 256) {
      int nn = idx / 28, cc = idx % 28;
      size_t row = (size_t)b * N_ + nb + nn;
      qq[nn][cc] = qb[row * 224 + h * 28 + cc];
      kk[nn][cc] = kvb[row * 448 + h * 28 + cc];
    }
    __syncthreads();
    #pragma unroll
    for (int u = 0; u < 4; ++u) {
      if (p[u] < 784) {
        int i = p[u] / 28, j = p[u] % 28;
        float s = 0.f;
        #pragma unroll
        for (int nn = 0; nn < 8; ++nn) s += kk[nn][i] * qq[nn][j];
        g[u] += s;
      }
    }
    if (tid < 28) { for (int nn = 0; nn < 8; ++nn) sq += kk[nn][tid] * kk[nn][tid]; }
    else if (tid >= 32 && tid < 60) {
      int c = tid - 32;
      for (int nn = 0; nn < 8; ++nn) sq += qq[nn][c] * qq[nn][c];
    }
    __syncthreads();
  }
  if (tid < 28) ksq[tid] = sq;
  if (tid >= 32 && tid < 60) qsq[tid - 32] = sq;
  #pragma unroll
  for (int u = 0; u < 4; ++u)
    if (p[u] < 784) S[p[u] / 28][p[u] % 28] = g[u];
  __syncthreads();
  for (int q2 = tid; q2 < 784; q2 += 256) {
    int i = q2 / 28, j = q2 % 28;
    float nk = fmaxf(sqrtf(ksq[i]), 1e-12f);
    float nq = fmaxf(sqrtf(qsq[j]), 1e-12f);
    S[i][j] = S[i][j] / (nk * nq) * scale;
  }
  __syncthreads();
  if (tid < 28) {
    float m = -1e30f;
    for (int j = 0; j < 28; ++j) m = fmaxf(m, S[tid][j]);
    float s = 0.f;
    for (int j = 0; j < 28; ++j) { float e = expf(S[tid][j] - m); S[tid][j] = e; s += e; }
    float inv = 1.f / s;
    for (int j = 0; j < 28; ++j) S[tid][j] *= inv;
  }
  __syncthreads();
  // xa[n,i] = sum_j S[i][j]*v[n,j];  out = xa + x
  for (int nb = 0; nb < N_; nb += 8) {
    for (int idx = tid; idx < 8 * 28; idx += 256) {
      int nn = idx / 28, cc = idx % 28;
      size_t row = (size_t)b * N_ + nb + nn;
      vv[nn][cc] = kvb[row * 448 + 224 + h * 28 + cc];
    }
    __syncthreads();
    for (int idx = tid; idx < 8 * 28; idx += 256) {
      int nn = idx / 28, i = idx % 28;
      float s = 0.f;
      #pragma unroll
      for (int j = 0; j < 28; ++j) s += S[i][j] * vv[nn][j];
      size_t gi = ((size_t)b * N_ + nb + nn) * 224 + h * 28 + i;
      xout[gi] = s + xin[gi];
    }
    __syncthreads();
  }
}

// ---------------------------------------------------------------- LayerNorm over d_model=32
// X: (B,H,W,DIM) -> token row r=b*L_+l, l=ww*DIM_+cc, channel m = hh
__global__ __launch_bounds__(256) void k_ln32(const float* __restrict__ X,
                                              const float* __restrict__ g,
                                              const float* __restrict__ bb,
                                              float* __restrict__ xn32,
                                              _Float16* __restrict__ xn16) {
  size_t r = (size_t)blockIdx.x * 256 + threadIdx.x;
  if (r >= (size_t)MROWS_) return;
  int b = (int)(r / L_), l = (int)(r % L_);
  int ww = l / DIM_, cc = l % DIM_;
  float v[32]; float m = 0.f;
  #pragma unroll
  for (int i = 0; i < 32; ++i) {
    v[i] = X[(((size_t)b * H_ + i) * W_ + ww) * DIM_ + cc];
    m += v[i];
  }
  m *= (1.f / 32.f);
  float var = 0.f;
  #pragma unroll
  for (int i = 0; i < 32; ++i) { float d = v[i] - m; var += d * d; }
  var *= (1.f / 32.f);
  float inv = rsqrtf(var + 1e-5f);
  #pragma unroll
  for (int i = 0; i < 32; ++i) {
    float o = (v[i] - m) * inv * g[i] + bb[i];
    xn32[r * 32 + i] = o;
    xn16[r * 32 + i] = (_Float16)o;
  }
}

// ---------------------------------------------------------------- causal depthwise conv1d + SiLU
__global__ __launch_bounds__(256) void k_conv1d(const float* __restrict__ xz,
                                                const float* __restrict__ cw,
                                                const float* __restrict__ cb,
                                                _Float16* __restrict__ xc16) {
  size_t idx = (size_t)blockIdx.x * 256 + threadIdx.x;
  if (idx >= (size_t)MROWS_ * DI_) return;
  int d = (int)(idx & 63);
  size_t r = idx >> 6;
  int b = (int)(r / L_), l = (int)(r % L_);
  float s = cb[d];
  #pragma unroll
  for (int j = 0; j < 4; ++j) {
    int lt = l - 3 + j;
    if (lt >= 0) s += xz[((size_t)b * L_ + lt) * 128 + d] * cw[d * 4 + j];
  }
  s = s / (1.f + expf(-s));   // SiLU
  xc16[r * 64 + d] = (_Float16)s;
}

// ---------------------------------------------------------------- selective scan
// one block per batch; tid = d*16 + n; 16-lane shuffle reduction over state n
__global__ __launch_bounds__(1024) void k_scan(const float* __restrict__ xdbl,
                                               const _Float16* __restrict__ xc16,
                                               const float* __restrict__ xz,
                                               const float* __restrict__ dtW,
                                               const float* __restrict__ dtb,
                                               const float* __restrict__ A_log,
                                               const float* __restrict__ Dp,
                                               _Float16* __restrict__ y16) {
  const int b = blockIdx.x;
  const int tid = threadIdx.x;
  const int d = tid >> 4, n = tid & 15;
  const float A   = -expf(A_log[d * 16 + n]);
  const float w0  = dtW[d];
  const float w1  = dtW[64 + d];
  const float db  = dtb[d];
  const float Dpd = Dp[d];
  float h = 0.f;
  for (int t = 0; t < L_; ++t) {
    size_t base = (size_t)b * L_ + t;
    float xd0 = xdbl[base * 34 + 0];
    float xd1 = xdbl[base * 34 + 1];
    float pre = xd0 * w0 + xd1 * w1 + db;
    float dt  = (pre > 20.f) ? pre : log1pf(expf(pre));   // softplus
    float Bv  = xdbl[base * 34 + 2 + n];
    float Cv  = xdbl[base * 34 + 18 + n];
    float u   = (float)xc16[base * 64 + d];
    h = expf(dt * A) * h + (dt * u) * Bv;
    float c = h * Cv;
    c += __shfl_xor(c, 1, 32);
    c += __shfl_xor(c, 2, 32);
    c += __shfl_xor(c, 4, 32);
    c += __shfl_xor(c, 8, 32);
    if (n == 0) {
      float y = c + u * Dpd;
      float z = xz[base * 128 + 64 + d];
      y *= z / (1.f + expf(-z));   // y * silu(z)
      y16[base * 64 + d] = (_Float16)y;
    }
  }
}

// ---------------------------------------------------------------- xm + skip*xn -> LN -> f16
__global__ __launch_bounds__(256) void k_skip_ln(const float* __restrict__ xm,
                                                 const float* __restrict__ xn32,
                                                 const float* __restrict__ skip,
                                                 const float* __restrict__ g,
                                                 const float* __restrict__ bb,
                                                 _Float16* __restrict__ out16) {
  size_t r = (size_t)blockIdx.x * 256 + threadIdx.x;
  if (r >= (size_t)MROWS_) return;
  const float sk = skip[0];
  float v[32]; float m = 0.f;
  #pragma unroll
  for (int i = 0; i < 32; ++i) { v[i] = xm[r * 32 + i] + sk * xn32[r * 32 + i]; m += v[i]; }
  m *= (1.f / 32.f);
  float var = 0.f;
  #pragma unroll
  for (int i = 0; i < 32; ++i) { float d = v[i] - m; var += d * d; }
  var *= (1.f / 32.f);
  float inv = rsqrtf(var + 1e-5f);
  #pragma unroll
  for (int i = 0; i < 32; ++i)
    out16[r * 32 + i] = (_Float16)((v[i] - m) * inv * g[i] + bb[i]);
}

// ---------------------------------------------------------------- residual back to (B,H,W,DIM)
__global__ __launch_bounds__(256) void k_resback(const float* __restrict__ XR,
                                                 const float* __restrict__ pr,
                                                 const float* __restrict__ pb,
                                                 float* __restrict__ X2) {
  size_t e = (size_t)blockIdx.x * 256 + threadIdx.x;
  if (e >= (size_t)B_ * H_ * W_ * DIM_) return;
  int cc = (int)(e % DIM_);
  size_t t = e / DIM_;
  int ww = (int)(t % W_); t /= W_;
  int hh = (int)(t % H_); int b = (int)(t / H_);
  size_t r = (size_t)b * L_ + ww * DIM_ + cc;
  X2[e] = XR[e] + pr[r * 32 + hh] + pb[hh];
}

// ---------------------------------------------------------------- 5x5x5 conv over (h,w,c)
__global__ __launch_bounds__(256) void k_conv3d(const float* __restrict__ X2,
                                                const float* __restrict__ w3,
                                                const float* __restrict__ b3,
                                                float* __restrict__ XC) {
  size_t e = (size_t)blockIdx.x * 256 + threadIdx.x;
  if (e >= (size_t)B_ * H_ * W_ * DIM_) return;
  int cc = (int)(e % DIM_);
  size_t t = e / DIM_;
  int ww = (int)(t % W_); t /= W_;
  int hh = (int)(t % H_); int b = (int)(t / H_);
  float acc = 0.f;
  for (int dz = 0; dz < 5; ++dz) {
    int zi = hh + dz - 2; if (zi < 0 || zi >= H_) continue;
    for (int dy = 0; dy < 5; ++dy) {
      int yi = ww + dy - 2; if (yi < 0 || yi >= W_) continue;
      const float* row = X2 + (((size_t)b * H_ + zi) * W_ + yi) * DIM_;
      const float* wr  = w3 + (dz * 5 + dy) * 5;
      #pragma unroll
      for (int dx = 0; dx < 5; ++dx) {
        int xi = cc + dx - 2; if (xi < 0 || xi >= DIM_) continue;
        acc += row[xi] * wr[dx];
      }
    }
  }
  XC[e] = acc + b3[0];
}

// ---------------------------------------------------------------- DWT -> padded window tokens (4 subbands, f16)
__global__ __launch_bounds__(256) void k_dwt_tokens(const float* __restrict__ XC,
                                                    _Float16* __restrict__ T16) {
  size_t idx = (size_t)blockIdx.x * 256 + threadIdx.x;
  if (idx >= (size_t)B_ * DIM_ * 16 * 24) return;
  int x = (int)(idx % 24);
  size_t t = idx / 24;
  int y = (int)(t % 16); t /= 16;
  int c = (int)(t % DIM_); int b = (int)(t / DIM_);
  int xs = (x < 20) ? x : (38 - x);   // reflect pad (edge excluded)
  size_t base = (((size_t)b * H_ + 2 * y) * W_ + 2 * xs) * DIM_ + c;
  float p00 = XC[base];
  float p01 = XC[base + DIM_];
  float p10 = XC[base + (size_t)W_ * DIM_];
  float p11 = XC[base + (size_t)W_ * DIM_ + DIM_];
  float lo0 = (p00 + p01) * RS_, lo1 = (p10 + p11) * RS_;
  float hi0 = (p00 - p01) * RS_, hi1 = (p10 - p11) * RS_;
  float cA = (lo0 + lo1) * RS_, cH = (lo0 - lo1) * RS_;
  float cV = (hi0 + hi1) * RS_, cD = (hi0 - hi1) * RS_;
  size_t row = ((size_t)b * 6 + (y >> 3) * 3 + (x >> 3)) * 64 + (y & 7) * 8 + (x & 7);
  const size_t sub = (size_t)TROWS_ * DIM_;
  T16[0 * sub + row * DIM_ + c] = (_Float16)cA;
  T16[1 * sub + row * DIM_ + c] = (_Float16)cH;
  T16[2 * sub + row * DIM_ + c] = (_Float16)cV;
  T16[3 * sub + row * DIM_ + c] = (_Float16)cD;
}

// ---------------------------------------------------------------- 64-token window attention core
__global__ __launch_bounds__(256) void k_winattn(const float* __restrict__ QW,
                                                 const float* __restrict__ KVW,
                                                 const float* __restrict__ pe,
                                                 _Float16* __restrict__ O16) {
  const int win = blockIdx.x >> 3, h = blockIdx.x & 7;
  const int tid = threadIdx.x;
  const float scale = 0.18898223650461363f;  // 28^-0.5
  __shared__ _Float16 qs[64 * 28], ks[64 * 28], vs[64 * 28];
  __shared__ float aS[64 * 65];

  for (int idx = tid; idx < 64 * 28; idx += 256) {
    int i = idx / 28, dd = idx % 28;
    size_t row = (size_t)win * 64 + i;
    qs[idx] = (_Float16)(QW[row * 224 + h * 28 + dd] * scale);
    ks[idx] = (_Float16)(KVW[row * 448 + h * 28 + dd]);
    vs[idx] = (_Float16)(KVW[row * 448 + 224 + h * 28 + dd]);
  }
  __syncthreads();
  for (int p2 = tid; p2 < 4096; p2 += 256) {
    int i = p2 >> 6, j = p2 & 63;
    float s = 0.f;
    #pragma unroll
    for (int dd = 0; dd < 28; ++dd) s += (float)qs[i * 28 + dd] * (float)ks[j * 28 + dd];
    aS[i * 65 + j] = s + pe[(size_t)h * 4096 + i * 64 + j];
  }
  __syncthreads();
  if (tid < 64) {
    float m = -1e30f;
    for (int j = 0; j < 64; ++j) m = fmaxf(m, aS[tid * 65 + j]);
    float s = 0.f;
    for (int j = 0; j < 64; ++j) { float e = expf(aS[tid * 65 + j] - m); aS[tid * 65 + j] = e; s += e; }
    float inv = 1.f / s;
    for (int j = 0; j < 64; ++j) aS[tid * 65 + j] *= inv;
  }
  __syncthreads();
  for (int p2 = tid; p2 < 64 * 28; p2 += 256) {
    int i = p2 / 28, dd = p2 % 28;
    float s = 0.f;
    #pragma unroll
    for (int j = 0; j < 64; ++j) s += aS[i * 65 + j] * (float)vs[j * 28 + dd];
    O16[((size_t)win * 64 + i) * 224 + h * 28 + dd] = (_Float16)s;
  }
}

// ---------------------------------------------------------------- inverse DWT + output
__global__ __launch_bounds__(256) void k_iwt_out(const float* __restrict__ OP,
                                                 const float* __restrict__ boA,
                                                 const float* __restrict__ boH,
                                                 const float* __restrict__ boV,
                                                 const float* __restrict__ boD,
                                                 float* __restrict__ out) {
  size_t e = (size_t)blockIdx.x * 256 + threadIdx.x;
  if (e >= (size_t)B_ * H_ * W_ * DIM_) return;
  int c = (int)(e % DIM_);
  size_t t = e / DIM_;
  int X = (int)(t % W_); t /= W_;
  int Y = (int)(t % H_); int b = (int)(t / H_);
  int y = Y >> 1, x = X >> 1;
  size_t row = ((size_t)b * 6 + (y >> 3) * 3 + (x >> 3)) * 64 + (y & 7) * 8 + (x & 7);
  const size_t sub = (size_t)TROWS_ * DIM_;
  float a  = OP[0 * sub + row * DIM_ + c] + boA[c];
  float hc = OP[1 * sub + row * DIM_ + c] + boH[c];
  float v  = OP[2 * sub + row * DIM_ + c] + boV[c];
  float dd = OP[3 * sub + row * DIM_ + c] + boD[c];
  float lo = ((Y & 1) ? (a - hc) : (a + hc)) * RS_;
  float hi = ((Y & 1) ? (v - dd) : (v + dd)) * RS_;
  float val = ((X & 1) ? (lo - hi) : (lo + hi)) * RS_;
  out[e] = val;
}

// ================================================================ launch
extern "C" void kernel_launch(void* const* d_in, const int* in_sizes, int n_in,
                              void* d_out, int out_size, void* d_ws, size_t ws_size,
                              hipStream_t stream) {
  const float* x         = (const float*)d_in[0];
  const float* Wq        = (const float*)d_in[1];
  const float* Wkv       = (const float*)d_in[2];
  const float* ln_g      = (const float*)d_in[3];
  const float* ln_b      = (const float*)d_in[4];
  const float* in_proj_W = (const float*)d_in[5];
  const float* conv1d_W  = (const float*)d_in[6];
  const float* conv1d_b  = (const float*)d_in[7];
  const float* x_proj_W  = (const float*)d_in[8];
  const float* dt_proj_W = (const float*)d_in[9];
  const float* dt_proj_b = (const float*)d_in[10];
  const float* A_log     = (const float*)d_in[11];
  const float* Dp        = (const float*)d_in[12];
  const float* out_proj_W= (const float*)d_in[13];
  const float* skip      = (const float*)d_in[14];
  const float* proj_W    = (const float*)d_in[15];
  const float* proj_b    = (const float*)d_in[16];
  const float* conv3d_W  = (const float*)d_in[17];
  const float* conv3d_b  = (const float*)d_in[18];
  const float* Wq1       = (const float*)d_in[19];
  const float* Wkv1      = (const float*)d_in[20];
  const float* pos_emb   = (const float*)d_in[21];
  const float* Wo[4]  = { (const float*)d_in[22], (const float*)d_in[24],
                          (const float*)d_in[26], (const float*)d_in[28] };
  const float* bo[4]  = { (const float*)d_in[23], (const float*)d_in[25],
                          (const float*)d_in[27], (const float*)d_in[29] };

  // ---- workspace layout (bump allocator, 256B aligned) ----
  char* wsp = (char*)d_ws;
  size_t off = 0;
  auto take = [&](size_t bytes) -> char* {
    char* p = wsp + off;
    off = (off + bytes + 255) & ~(size_t)255;
    return p;
  };
  _Float16* x16    = (_Float16*)take((size_t)QROWS_ * DIM_ * 2);
  _Float16* Wq16   = (_Float16*)take(224 * 224 * 2);
  _Float16* Wkv16  = (_Float16*)take(224 * 448 * 2);
  _Float16* Wq116  = (_Float16*)take(224 * 224 * 2);
  _Float16* Wkv116 = (_Float16*)take(224 * 448 * 2);
  _Float16* Wo16[4];
  for (int s = 0; s < 4; ++s) Wo16[s] = (_Float16*)take(224 * 224 * 2);
  _Float16* inW16  = (_Float16*)take(32 * 128 * 2);
  _Float16* xpW16  = (_Float16*)take(64 * 34 * 2);
  _Float16* outW16 = (_Float16*)take(64 * 32 * 2);
  _Float16* prW16  = (_Float16*)take(32 * 32 * 2);
  float* qbuf  = (float*)take((size_t)QROWS_ * 224 * 4);     // later: pr
  float* kvbuf = (float*)take((size_t)QROWS_ * 448 * 4);     // later: X2 | XC
  float* XR    = (float*)take((size_t)QROWS_ * 224 * 4);     // x + channel attention
  float* xn32  = (float*)take((size_t)MROWS_ * 32 * 4);
  _Float16* xn16 = (_Float16*)take((size_t)MROWS_ * 32 * 2);
  float* xz    = (float*)take((size_t)MROWS_ * 128 * 4);     // later: wavelet token region
  _Float16* xc16 = (_Float16*)take((size_t)MROWS_ * 64 * 2);
  float* xdbl  = (float*)take((size_t)MROWS_ * 34 * 4);
  _Float16* y16 = (_Float16*)take((size_t)MROWS_ * 64 * 2);
  float* xm    = (float*)take((size_t)MROWS_ * 32 * 4);
  _Float16* xmn16 = (_Float16*)take((size_t)MROWS_ * 32 * 2);

  // aliases (lifetimes do not overlap)
  float* pr = qbuf;
  float* X2 = kvbuf;
  float* XC = kvbuf + (size_t)QROWS_ * 224;
  char* rg = (char*)xz;
  _Float16* T16 = (_Float16*)rg;                                           // 4 subbands
  float* QW  = (float*)(rg + (size_t)4 * TROWS_ * 224 * 2);
  float* KVW = QW + (size_t)TROWS_ * 224;
  _Float16* O16 = (_Float16*)((char*)(KVW + (size_t)TROWS_ * 448));
  float* OP  = (float*)((char*)O16 + (size_t)TROWS_ * 224 * 2);            // 4 subbands f32

  auto cast = [&](const float* s, _Float16* d, int n) {
    k_cast_f16<<<(n + 255) / 256, 256, 0, stream>>>(s, d, n);
  };
  auto gemm = [&](const _Float16* A, const _Float16* Bm, float* C,
                  int M, int N, int K, int ldb, int ldc) {
    dim3 grid((N + 63) / 64, M / 128);
    k_gemm<<<grid, 128, 0, stream>>>(A, Bm, C, M, N, K, ldb, ldc);
  };

  // ---- 0) f16 casts ----
  cast(x, x16, QROWS_ * DIM_);
  cast(Wq, Wq16, 224 * 224);
  cast(Wkv, Wkv16, 224 * 448);
  cast(Wq1, Wq116, 224 * 224);
  cast(Wkv1, Wkv116, 224 * 448);
  for (int s = 0; s < 4; ++s) cast(Wo[s], Wo16[s], 224 * 224);
  cast(in_proj_W, inW16, 32 * 128);
  cast(x_proj_W, xpW16, 64 * 34);
  cast(out_proj_W, outW16, 64 * 32);
  cast(proj_W, prW16, 32 * 32);

  // ---- 1) channel attention ----
  gemm(x16, Wq16,  qbuf,  QROWS_, 224, 224, 224, 224);
  gemm(x16, Wkv16, kvbuf, QROWS_, 448, 224, 448, 448);
  k_chan_attn<<<B_ * HEADS_, 256, 0, stream>>>(qbuf, kvbuf, x, XR);

  // ---- 2) mamba ----
  k_ln32<<<(MROWS_ + 255) / 256, 256, 0, stream>>>(XR, ln_g, ln_b, xn32, xn16);
  gemm(xn16, inW16, xz, MROWS_, 128, 32, 128, 128);
  {
    size_t n = (size_t)MROWS_ * DI_;
    k_conv1d<<<(unsigned)((n + 255) / 256), 256, 0, stream>>>(xz, conv1d_W, conv1d_b, xc16);
  }
  gemm(xc16, xpW16, xdbl, MROWS_, 34, 64, 34, 34);
  k_scan<<<B_, 1024, 0, stream>>>(xdbl, xc16, xz, dt_proj_W, dt_proj_b, A_log, Dp, y16);
  gemm(y16, outW16, xm, MROWS_, 32, 64, 32, 32);
  k_skip_ln<<<(MROWS_ + 255) / 256, 256, 0, stream>>>(xm, xn32, skip, ln_g, ln_b, xmn16);
  gemm(xmn16, prW16, pr, MROWS_, 32, 32, 32, 32);

  const size_t NEL = (size_t)B_ * H_ * W_ * DIM_;
  k_resback<<<(unsigned)((NEL + 255) / 256), 256, 0, stream>>>(XR, pr, proj_b, X2);

  // ---- 3) conv3d ----
  k_conv3d<<<(unsigned)((NEL + 255) / 256), 256, 0, stream>>>(X2, conv3d_W, conv3d_b, XC);

  // ---- 4) DWT + window attention (4 subbands) + IWT ----
  {
    size_t n = (size_t)B_ * DIM_ * 16 * 24;
    k_dwt_tokens<<<(unsigned)((n + 255) / 256), 256, 0, stream>>>(XC, T16);
  }
  for (int s = 0; s < 4; ++s) {
    const _Float16* Ts = T16 + (size_t)s * TROWS_ * 224;
    gemm(Ts, Wq116,  QW,  TROWS_, 224, 224, 224, 224);
    gemm(Ts, Wkv116, KVW, TROWS_, 448, 224, 448, 448);
    k_winattn<<<NWIN_ * HEADS_, 256, 0, stream>>>(QW, KVW, pos_emb, O16);
    gemm(O16, Wo16[s], OP + (size_t)s * TROWS_ * 224, TROWS_, 224, 224, 224, 224);
  }
  k_iwt_out<<<(unsigned)((NEL + 255) / 256), 256, 0, stream>>>(
      OP, bo[0], bo[1], bo[2], bo[3], (float*)d_out);
}
